// BitLayer_70677981823568
// MI455X (gfx1250) — 8-bit matrix operand order repaired and checked by reading, compiled, not run on hardware
//
#include <hip/hip_runtime.h>
#include <hip/hip_bf16.h>

#define B_ 32
#define I_ 512
#define O_ 512
#define L_ 256

typedef int            v8i   __attribute__((ext_vector_type(8)));
typedef unsigned short u16x4 __attribute__((ext_vector_type(4)));
typedef unsigned int   u32x4 __attribute__((ext_vector_type(4)));
typedef int            v4ig  __attribute__((vector_size(16)));

#define AS1 __attribute__((address_space(1)))
#define AS3 __attribute__((address_space(3)))

#if __has_builtin(__builtin_amdgcn_global_load_async_to_lds_b128)
#define HAVE_ASYNC_LDS 1
#else
#define HAVE_ASYNC_LDS 0
#endif

__device__ __forceinline__ void wait_asynccnt0() {
#if __has_builtin(__builtin_amdgcn_s_wait_asynccnt)
  __builtin_amdgcn_s_wait_asynccnt(0);
#else
  asm volatile("s_wait_asynccnt 0" ::: "memory");
#endif
}

__global__ __launch_bounds__(256) void prep_n(const float* __restrict__ kern,
                                              unsigned short* __restrict__ N) {
  int t = blockIdx.x * blockDim.x + threadIdx.x;
  if (t < O_ * I_) {
    float v = fminf(fmaxf(kern[t], 0.0f), 1.0f);
    float r = rintf(v * 256.0f);
    N[t] = (unsigned short)(int)r;
  }
}

__global__ __launch_bounds__(256) void prep_x(const int* __restrict__ x,
                                              unsigned char* __restrict__ X) {
  const int b = blockIdx.x;
  const int j = threadIdx.x;
  unsigned char* dst = X + ((size_t)j * B_ + b) * I_;
  const int* src = x + (size_t)b * I_ * L_ + j;
  for (int i = 0; i < I_; i += 4) {
    unsigned int p = ((unsigned int)(src[(size_t)(i + 0) * L_] & 1))
                   | ((unsigned int)(src[(size_t)(i + 1) * L_] & 1) << 8)
                   | ((unsigned int)(src[(size_t)(i + 2) * L_] & 1) << 16)
                   | ((unsigned int)(src[(size_t)(i + 3) * L_] & 1) << 24);
    *(unsigned int*)(dst + i) = p;
  }
}

__global__ __launch_bounds__(256) void bitlayer_wmma(
    const unsigned char* __restrict__ X,
    const unsigned short* __restrict__ N,
    float* __restrict__ out)
{
  const int j    = blockIdx.x;
  const int wave = threadIdx.x >> 5;
  const int lane = threadIdx.x & 31;
  const int m    = lane & 15;
  const int hi   = lane >> 4;
  const int khA  = hi * 8;
  const int khB  = hi * 16;

  __shared__ __align__(16) unsigned char sX[B_ * I_];
  const unsigned char* Xj = X + (size_t)j * B_ * I_;
  {
    const unsigned char* g = Xj + (size_t)threadIdx.x * 64;
    unsigned char*       l = sX + (size_t)threadIdx.x * 64;
#if HAVE_ASYNC_LDS
    __builtin_amdgcn_global_load_async_to_lds_b128((AS1 v4ig*)g, (AS3 v4ig*)l, 0, 0);
    __builtin_amdgcn_global_load_async_to_lds_b128((AS1 v4ig*)g, (AS3 v4ig*)l, 16, 0);
    __builtin_amdgcn_global_load_async_to_lds_b128((AS1 v4ig*)g, (AS3 v4ig*)l, 32, 0);
    __builtin_amdgcn_global_load_async_to_lds_b128((AS1 v4ig*)g, (AS3 v4ig*)l, 48, 0);
    wait_asynccnt0();
#else
    const u32x4* gs = (const u32x4*)g;
    u32x4*       ls = (u32x4*)l;
    ls[0] = gs[0]; ls[1] = gs[1]; ls[2] = gs[2]; ls[3] = gs[3];
#endif
  }
  __syncthreads();

  const int ot = blockIdx.y * 8 + wave;
  const unsigned short* Nrow = N + (size_t)(ot * 16 + m) * I_;
  __builtin_prefetch(Nrow, 0, 3);

  for (int bt = 0; bt < 2; ++bt) {
    const unsigned char* Xrow = sX + (size_t)(bt * 16 + m) * I_;
    v8i acc = {};
    #pragma unroll
    for (int kk = 0; kk < I_; kk += 64) {
      v8i a;
      #pragma unroll
      for (int v = 0; v < 8; ++v) {
        const int kA = kk + khA + ((v & 1) * 4) + (((v >> 1) & 1) * 16)
                                + (((v >> 2) & 1) * 32);
        a[v] = (int)*(const unsigned int*)(Xrow + kA);
      }
      v8i b;
      #pragma unroll
      for (int v = 0; v < 8; ++v) {
        const int kB = kk + khA + ((v & 1) * 4) + (((v >> 1) & 1) * 16) + (((v >> 2) & 1) * 32);
        u16x4 s = *(const u16x4*)(Nrow + kB);
        unsigned int w = ((unsigned int)(j < (int)s.x))
                       | (((unsigned int)(j < (int)s.y)) << 8)
                       | (((unsigned int)(j < (int)s.z)) << 16)
                       | (((unsigned int)(j < (int)s.w)) << 24);
        b[v] = (int)w;
      }
      acc = __builtin_amdgcn_wmma_i32_16x16x64_iu8(
           false, a,  false, b, acc,
           false,  false);
    }
    #pragma unroll
    for (int v = 0; v < 8; ++v) {
      const int bb = bt * 16 + (v + hi * 8);
      const int oo = ot * 16 + m;
      out[((size_t)bb * O_ + oo) * L_ + j] = (acc[v] > 0) ? 1.0f : 0.0f;
    }
  }
}

extern "C" void kernel_launch(void* const* d_in, const int* in_sizes, int n_in,
                              void* d_out, int out_size, void* d_ws, size_t ws_size,
                              hipStream_t stream) {
  const int*   x    = (const int*)d_in[0];
  const float* kern = (const float*)d_in[1];
  float*       out  = (float*)d_out;

  unsigned short* N  = (unsigned short*)d_ws;
  unsigned char*  Xp = (unsigned char*)d_ws + (size_t)O_ * I_ * sizeof(unsigned short);

  prep_n<<<(O_ * I_ + 255) / 256, 256, 0, stream>>>(kern, N);
  prep_x<<<B_, L_, 0, stream>>>(x, Xp);
  bitlayer_wmma<<<dim3(L_, 4), 256, 0, stream>>>(Xp, N, out);
}
